// PrunableMixtralSparseMoeBlockWrapper_2010044694834
// MI455X (gfx1250) — compile-verified, matrix operand order checked by reading, not run on hardware
//
#include <hip/hip_runtime.h>
#include <hip/hip_bf16.h>

// ---- problem constants (fixed by the reference) ----
#define T_TOK 4096   // B*S = 2*2048
#define DDIM  1024
#define FDIM  3584
#define NEXP  8

typedef __attribute__((ext_vector_type(16))) __bf16 v16bf;
typedef __attribute__((ext_vector_type(8)))  __bf16 v8bf;
typedef __attribute__((ext_vector_type(8)))  float  v8f;

// ---------------------------------------------------------------------------
// WMMA: D = A(16x32 bf16) * B(32x16 bf16) + C(16x16 f32)
// ---------------------------------------------------------------------------
__device__ __forceinline__ v8f wmma_bf16(v16bf a, v16bf b, v8f c) {
  return __builtin_amdgcn_wmma_f32_16x16x32_bf16(
      /*neg_a=*/false, a, /*neg_b=*/false, b,
      /*c_mod=*/(short)0, c, /*reuse_a=*/false, /*reuse_b=*/false);
}

// Fragment loaders. Lane layout (ISA 7.12.2, 16-bit 16x32): lane<16 holds
// K={0..7,16..23}, lane>=16 holds K={8..15,24..31}. Caller folds the
// +8*(lane>>4) K-offset into the row pointer.

// bf16 source: two 16B loads, no conversion (hot path). Scale ignored.
__device__ __forceinline__ v16bf frag_load(const __bf16* __restrict__ p,
                                           float /*scale*/) {
  v8bf lo = *(const v8bf*)(p + 0);
  v8bf hi = *(const v8bf*)(p + 16);
  v16bf r;
#pragma unroll
  for (int i = 0; i < 8; ++i) { r[i] = lo[i]; r[i + 8] = hi[i]; }
  return r;
}

// fp32 source: four 16B loads + in-register cvt (fallback path).
__device__ __forceinline__ v16bf frag_load(const float* __restrict__ p,
                                           float scale) {
  float4 a0 = *(const float4*)(p + 0);
  float4 a1 = *(const float4*)(p + 4);
  float4 a2 = *(const float4*)(p + 16);
  float4 a3 = *(const float4*)(p + 20);
  v16bf r;
  r[0]  = (__bf16)(a0.x * scale); r[1]  = (__bf16)(a0.y * scale);
  r[2]  = (__bf16)(a0.z * scale); r[3]  = (__bf16)(a0.w * scale);
  r[4]  = (__bf16)(a1.x * scale); r[5]  = (__bf16)(a1.y * scale);
  r[6]  = (__bf16)(a1.z * scale); r[7]  = (__bf16)(a1.w * scale);
  r[8]  = (__bf16)(a2.x * scale); r[9]  = (__bf16)(a2.y * scale);
  r[10] = (__bf16)(a2.z * scale); r[11] = (__bf16)(a2.w * scale);
  r[12] = (__bf16)(a3.x * scale); r[13] = (__bf16)(a3.y * scale);
  r[14] = (__bf16)(a3.z * scale); r[15] = (__bf16)(a3.w * scale);
  return r;
}

// ---------------------------------------------------------------------------
// Router: logits = x @ gate_w^T, softmax, top-2, normalized combine weights.
// ---------------------------------------------------------------------------
__global__ __launch_bounds__(256) void moe_router(
    const float* __restrict__ x, const float* __restrict__ gw,
    float* __restrict__ rlog, float* __restrict__ wfull) {
  const int t = blockIdx.x * blockDim.x + threadIdx.x;
  if (t >= T_TOK) return;
  const float* xr = x + (size_t)t * DDIM;

  float acc[NEXP];
#pragma unroll
  for (int e = 0; e < NEXP; ++e) acc[e] = 0.f;

  for (int k = 0; k < DDIM; k += 4) {
    float4 xv = *(const float4*)(xr + k);
#pragma unroll
    for (int e = 0; e < NEXP; ++e) {
      float4 gv = *(const float4*)(gw + (size_t)e * DDIM + k);
      acc[e] += xv.x * gv.x + xv.y * gv.y + xv.z * gv.z + xv.w * gv.w;
    }
  }

  float m = acc[0];
#pragma unroll
  for (int e = 1; e < NEXP; ++e) m = fmaxf(m, acc[e]);
  float ex[NEXP];
#pragma unroll
  for (int e = 0; e < NEXP; ++e) ex[e] = __expf(acc[e] - m);

  int i1 = 0;
#pragma unroll
  for (int e = 1; e < NEXP; ++e) if (ex[e] > ex[i1]) i1 = e;
  int i2 = (i1 == 0) ? 1 : 0;
#pragma unroll
  for (int e = 0; e < NEXP; ++e) if (e != i1 && e != i2 && ex[e] > ex[i2]) i2 = e;
  const float inv = 1.f / (ex[i1] + ex[i2]);

#pragma unroll
  for (int e = 0; e < NEXP; ++e) {
    rlog[(size_t)t * NEXP + e] = acc[e];
    float w = (e == i1) ? ex[i1] * inv : ((e == i2) ? ex[i2] * inv : 0.f);
    wfull[(size_t)t * NEXP + e] = w;
  }
}

// ---------------------------------------------------------------------------
// fp32 -> bf16 bulk convert (weights). 8 elements / thread, n % 8 == 0.
// ---------------------------------------------------------------------------
__global__ __launch_bounds__(256) void cvt_f32_bf16(
    const float* __restrict__ src, __bf16* __restrict__ dst, size_t n) {
  size_t i = ((size_t)blockIdx.x * blockDim.x + threadIdx.x) * 8;
  if (i >= n) return;
  float4 a = *(const float4*)(src + i);
  float4 b = *(const float4*)(src + i + 4);
  v8bf r;
  r[0] = (__bf16)a.x; r[1] = (__bf16)a.y; r[2] = (__bf16)a.z; r[3] = (__bf16)a.w;
  r[4] = (__bf16)b.x; r[5] = (__bf16)b.y; r[6] = (__bf16)b.z; r[7] = (__bf16)b.w;
  *(v8bf*)(dst + i) = r;
}

// Per-expert scaled activations: xs_b[t,d] = bf16(x[t,d] * wfull[t,e]).
__global__ __launch_bounds__(256) void moe_scale_x(
    const float* __restrict__ x, const float* __restrict__ wfull,
    __bf16* __restrict__ xsb, int e) {
  size_t i = ((size_t)blockIdx.x * blockDim.x + threadIdx.x) * 8;
  if (i >= (size_t)T_TOK * DDIM) return;
  const int t = (int)(i / DDIM);                 // DDIM % 8 == 0: single row
  const float wt = wfull[(size_t)t * NEXP + e];
  float4 a = *(const float4*)(x + i);
  float4 b = *(const float4*)(x + i + 4);
  v8bf r;
  r[0] = (__bf16)(a.x * wt); r[1] = (__bf16)(a.y * wt);
  r[2] = (__bf16)(a.z * wt); r[3] = (__bf16)(a.w * wt);
  r[4] = (__bf16)(b.x * wt); r[5] = (__bf16)(b.y * wt);
  r[6] = (__bf16)(b.z * wt); r[7] = (__bf16)(b.w * wt);
  *(v8bf*)(xsb + i) = r;
}

// ---------------------------------------------------------------------------
// GEMM1 (fused SwiGLU): h = silu(xs @ w1^T) * (xs @ w3^T)
// Block 256 thr = 8 waves as 4(M) x 2(N); wave tile 32(M) x 32(N) x {w1,w3}.
// Per K-step(32): 8 WMMAs, A frags reused x2(N), B frags reused x2(M).
// AT: activation elem type (bf16 pre-scaled | f32 + on-the-fly scale)
// WT: weight elem type (bf16 pre-converted | f32 on-the-fly cvt)
// ---------------------------------------------------------------------------
template <typename AT, typename WT>
__global__ __launch_bounds__(256) void moe_gemm1(
    const AT* __restrict__ a, const float* __restrict__ wfull,
    const WT* __restrict__ w1e, const WT* __restrict__ w3e,
    __bf16* __restrict__ h, int e) {
  const int wave  = threadIdx.x >> 5;
  const int mw    = wave & 3;          // 0..3
  const int nw    = wave >> 2;         // 0..1
  const int lane  = threadIdx.x & 31;
  const int lhalf = lane >> 4;
  const int l16   = lane & 15;

  const int m0 = blockIdx.x * 128 + mw * 32;
  const int n0 = blockIdx.y * 64  + nw * 32;
  const int kofs = lhalf * 8;          // per-lane K offset within fragment

  const int t0 = m0 + l16, t1 = m0 + 16 + l16;
  const float s0 = wfull[(size_t)t0 * NEXP + e];   // ignored for bf16 AT
  const float s1 = wfull[(size_t)t1 * NEXP + e];

  const AT* a0p = a + (size_t)t0 * DDIM + kofs;
  const AT* a1p = a + (size_t)t1 * DDIM + kofs;
  const WT* b10 = w1e + (size_t)(n0 + l16)      * DDIM + kofs;
  const WT* b11 = w1e + (size_t)(n0 + 16 + l16) * DDIM + kofs;
  const WT* b30 = w3e + (size_t)(n0 + l16)      * DDIM + kofs;
  const WT* b31 = w3e + (size_t)(n0 + 16 + l16) * DDIM + kofs;

  v8f acc1[2][2], acc3[2][2];
#pragma unroll
  for (int i = 0; i < 2; ++i)
#pragma unroll
    for (int j = 0; j < 2; ++j) { acc1[i][j] = (v8f)0.f; acc3[i][j] = (v8f)0.f; }

  for (int k = 0; k < DDIM; k += 32) {
    v16bf af0 = frag_load(a0p + k, s0);
    v16bf af1 = frag_load(a1p + k, s1);
    v16bf b;
    b = frag_load(b10 + k, 1.f);
    acc1[0][0] = wmma_bf16(af0, b, acc1[0][0]);
    acc1[1][0] = wmma_bf16(af1, b, acc1[1][0]);
    b = frag_load(b11 + k, 1.f);
    acc1[0][1] = wmma_bf16(af0, b, acc1[0][1]);
    acc1[1][1] = wmma_bf16(af1, b, acc1[1][1]);
    b = frag_load(b30 + k, 1.f);
    acc3[0][0] = wmma_bf16(af0, b, acc3[0][0]);
    acc3[1][0] = wmma_bf16(af1, b, acc3[1][0]);
    b = frag_load(b31 + k, 1.f);
    acc3[0][1] = wmma_bf16(af0, b, acc3[0][1]);
    acc3[1][1] = wmma_bf16(af1, b, acc3[1][1]);
  }

#pragma unroll
  for (int mf = 0; mf < 2; ++mf)
#pragma unroll
    for (int nf = 0; nf < 2; ++nf) {
      const int f = n0 + nf * 16 + l16;
#pragma unroll
      for (int r = 0; r < 8; ++r) {
        const int tt = m0 + mf * 16 + r + 8 * lhalf;  // C/D: row = r + 8*(lane>>4)
        const float g = acc1[mf][nf][r];
        const float v = (g / (1.f + __expf(-g))) * acc3[mf][nf][r];
        h[(size_t)tt * FDIM + f] = (__bf16)v;
      }
    }
}

// ---------------------------------------------------------------------------
// GEMM2: out (+)= h @ w2^T  (K=F). Block 8 waves as 4(M) x 2(N);
// wave tile 32(M) x 64(N): 8 WMMAs per K-step, A reused x4, B reused x2.
// ---------------------------------------------------------------------------
template <typename WT>
__global__ __launch_bounds__(256) void moe_gemm2(
    const __bf16* __restrict__ h, const WT* __restrict__ w2e,
    float* __restrict__ out, int beta) {
  const int wave  = threadIdx.x >> 5;
  const int mw    = wave & 3;
  const int nw    = wave >> 2;
  const int lane  = threadIdx.x & 31;
  const int lhalf = lane >> 4;
  const int l16   = lane & 15;

  const int m0 = blockIdx.x * 128 + mw * 32;
  const int n0 = blockIdx.y * 128 + nw * 64;
  const int kofs = lhalf * 8;

  const __bf16* a0p = h + (size_t)(m0 + l16)      * FDIM + kofs;
  const __bf16* a1p = h + (size_t)(m0 + 16 + l16) * FDIM + kofs;
  const WT* bp[4];
#pragma unroll
  for (int s = 0; s < 4; ++s)
    bp[s] = w2e + (size_t)(n0 + s * 16 + l16) * FDIM + kofs;

  v8f acc[2][4];
#pragma unroll
  for (int i = 0; i < 2; ++i)
#pragma unroll
    for (int s = 0; s < 4; ++s) acc[i][s] = (v8f)0.f;

  for (int k = 0; k < FDIM; k += 32) {
    v16bf af0 = frag_load(a0p + k, 1.f);
    v16bf af1 = frag_load(a1p + k, 1.f);
#pragma unroll
    for (int s = 0; s < 4; ++s) {
      v16bf b = frag_load(bp[s] + k, 1.f);
      acc[0][s] = wmma_bf16(af0, b, acc[0][s]);
      acc[1][s] = wmma_bf16(af1, b, acc[1][s]);
    }
  }

#pragma unroll
  for (int mf = 0; mf < 2; ++mf)
#pragma unroll
    for (int s = 0; s < 4; ++s) {
      const int d = n0 + s * 16 + l16;
#pragma unroll
      for (int r = 0; r < 8; ++r) {
        const int tt = m0 + mf * 16 + r + 8 * lhalf;
        const size_t idx = (size_t)tt * DDIM + d;
        const float v = acc[mf][s][r];
        out[idx] = beta ? (out[idx] + v) : v;   // beta=0 initializes d_out
      }
    }
}

// ---------------------------------------------------------------------------
extern "C" void kernel_launch(void* const* d_in, const int* in_sizes, int n_in,
                              void* d_out, int out_size, void* d_ws, size_t ws_size,
                              hipStream_t stream) {
  const float* x      = (const float*)d_in[0];  // [T, D]
  const float* gate_w = (const float*)d_in[1];  // [E, D]
  const float* w1     = (const float*)d_in[2];  // [E, F, D]
  const float* w3     = (const float*)d_in[3];  // [E, F, D]
  const float* w2     = (const float*)d_in[4];  // [E, D, F]

  float* out  = (float*)d_out;
  float* rlog = out + (size_t)T_TOK * DDIM;

  const size_t WELEM = (size_t)NEXP * FDIM * DDIM;   // per weight tensor

  // workspace layout
  char*   p     = (char*)d_ws;
  float*  wfull = (float*)p;              p += (size_t)T_TOK * NEXP * sizeof(float);
  __bf16* h     = (__bf16*)p;             p += (size_t)T_TOK * FDIM * 2;
  size_t  need_small = (size_t)(p - (char*)d_ws);
  __bf16* xsb   = (__bf16*)p;             p += (size_t)T_TOK * DDIM * 2;
  __bf16* w1b   = (__bf16*)p;             p += WELEM * 2;
  __bf16* w3b   = (__bf16*)p;             p += WELEM * 2;
  __bf16* w2b   = (__bf16*)p;             p += WELEM * 2;
  size_t  need_full = (size_t)(p - (char*)d_ws);
  (void)need_small;

  moe_router<<<T_TOK / 256, 256, 0, stream>>>(x, gate_w, rlog, wfull);

  if (ws_size >= need_full) {
    // Fast path: pre-convert weights to bf16 once (~23us of HBM traffic),
    // keeping GEMM inner loops conversion-free (pure b128 loads + WMMA).
    const int cvb = (int)((WELEM / 8 + 255) / 256);
    cvt_f32_bf16<<<cvb, 256, 0, stream>>>(w1, w1b, WELEM);
    cvt_f32_bf16<<<cvb, 256, 0, stream>>>(w3, w3b, WELEM);
    cvt_f32_bf16<<<cvb, 256, 0, stream>>>(w2, w2b, WELEM);

    for (int e = 0; e < NEXP; ++e) {
      const __bf16* w1e = w1b + (size_t)e * FDIM * DDIM;
      const __bf16* w3e = w3b + (size_t)e * FDIM * DDIM;
      const __bf16* w2e = w2b + (size_t)e * DDIM * FDIM;
      moe_scale_x<<<(T_TOK * DDIM / 8) / 256, 256, 0, stream>>>(x, wfull, xsb, e);
      moe_gemm1<__bf16, __bf16><<<dim3(T_TOK / 128, FDIM / 64), 256, 0, stream>>>(
          xsb, wfull, w1e, w3e, h, e);
      moe_gemm2<__bf16><<<dim3(T_TOK / 128, DDIM / 128), 256, 0, stream>>>(
          h, w2e, out, e != 0);
    }
  } else {
    // Fallback: convert fp32 -> bf16 on the fly inside the GEMMs.
    for (int e = 0; e < NEXP; ++e) {
      const float* w1e = w1 + (size_t)e * FDIM * DDIM;
      const float* w3e = w3 + (size_t)e * FDIM * DDIM;
      const float* w2e = w2 + (size_t)e * DDIM * FDIM;
      moe_gemm1<float, float><<<dim3(T_TOK / 128, FDIM / 64), 256, 0, stream>>>(
          x, wfull, w1e, w3e, h, e);
      moe_gemm2<float><<<dim3(T_TOK / 128, DDIM / 128), 256, 0, stream>>>(
          h, w2e, out, e != 0);
    }
  }
}
